// GlobalCrossAttention_52879637348578
// MI455X (gfx1250) — matrix operand order repaired and checked by reading, not compiled, not run on hardware
//
#include <hip/hip_runtime.h>

// ---------------------------------------------------------------------------
// Global cross attention with separable RPE, bf16 WMMA, flash-style softmax,
// split-K attention + combine.  gfx1250 / wave32 only.
//
// Layout choices (all feed WMMA fragments with contiguous b128 loads):
//   activations  : row-major [M][K]           -> A fragments contiguous
//   weights      : transposed bf16 [N][K]     -> B fragments contiguous
//   K            : [bh][key][d] (d = WMMA K)  -> K^T B fragments contiguous
//   V            : transposed [bh][d][key]    -> P.V B fragments contiguous
// GEMM k-loops are double-buffered so loads stay in flight under the WMMAs.
// ---------------------------------------------------------------------------

#define B_        2
#define NQ_       300
#define C_        256
#define NH_       8
#define HD_       32
#define H_        64
#define W_        64
#define HW_       4096
#define RPE_HID_  512
#define STRIDE_   16.0f
#define NQPAD_    304          // NQ padded to a multiple of 16
#define NSPLIT_   8
#define KEYS_PER_SPLIT_ (HW_ / NSPLIT_)   // 512

typedef __bf16 bf16_t;
typedef bf16_t bf16x16 __attribute__((ext_vector_type(16)));
typedef bf16_t bf16x8  __attribute__((ext_vector_type(8)));
typedef float  f32x8   __attribute__((ext_vector_type(8)));

static __device__ __forceinline__ bf16_t f2bf(float f) {
    unsigned u = __builtin_bit_cast(unsigned, f);
    unsigned r = u + 0x7FFFu + ((u >> 16) & 1u);           // round-to-nearest-even
    return __builtin_bit_cast(bf16_t, (unsigned short)(r >> 16));
}

static __device__ __forceinline__ f32x8 wmma_bf16(bf16x16 a, bf16x16 b, f32x8 c) {
    return __builtin_amdgcn_wmma_f32_16x16x32_bf16(
        /*neg_a=*/false, a, /*neg_b=*/false, b,
        /*c_mod=*/(short)0, c, /*reuse_a=*/false, /*reuse_b=*/false);
}

// A fragment: 16x32 bf16 tile at (row, k0) from row-major [*, lda].
// Per ISA: lane m = lane&15; K(e) = (e<8 ? e : e+8) + 8*(lane>>4).
static __device__ __forceinline__ bf16x16
load_a_frag(const bf16_t* __restrict__ p, int row, int lda, int k0) {
    int lane = threadIdx.x & 31;
    const bf16_t* base = p + (size_t)row * lda + k0 + ((lane >> 4) << 3);
    bf16x8 lo = *(const bf16x8*)(base);
    bf16x8 hi = *(const bf16x8*)(base + 16);
    bf16x16 a;
#pragma unroll
    for (int e = 0; e < 8; ++e) { a[e] = lo[e]; a[8 + e] = hi[e]; }
    return a;
}

// B fragment: 32x16 tile, source stored K-contiguous per N ("transposed",
// i.e. element (k, n) at p[n*ldn + k]).  Fully contiguous b128 loads.
// Per ISA: n = lane&15; K(e) = (e<8 ? e : e+8) + 8*(lane>>4).
static __device__ __forceinline__ bf16x16
load_b_frag_kcont(const bf16_t* __restrict__ p, int k0, int n0, int ldn) {
    int lane = threadIdx.x & 31;
    const bf16_t* base = p + (size_t)(n0 + (lane & 15)) * ldn + k0 + ((lane >> 4) << 3);
    bf16x8 lo = *(const bf16x8*)(base);
    bf16x8 hi = *(const bf16x8*)(base + 16);
    bf16x16 b;
#pragma unroll
    for (int e = 0; e < 8; ++e) { b[e] = lo[e]; b[8 + e] = hi[e]; }
    return b;
}

// Shared 16x64 GEMM core: A(16 rows starting at row) x Wt -> 4 x 16x16 acc.
// Double-buffered over the K dimension (C_/32 = 8 steps).
static __device__ __forceinline__ void
gemm_16x64(const bf16_t* __restrict__ A, int row,
           const bf16_t* __restrict__ Wt, int n0, f32x8 acc[4]) {
#pragma unroll
    for (int t = 0; t < 4; ++t)
#pragma unroll
        for (int j = 0; j < 8; ++j) acc[t][j] = 0.0f;

    bf16x16 a_cur = load_a_frag(A, row, C_, 0);
    bf16x16 b_cur[4];
#pragma unroll
    for (int t = 0; t < 4; ++t) b_cur[t] = load_b_frag_kcont(Wt, 0, n0 + 16 * t, C_);

#pragma unroll
    for (int k0 = 0; k0 < C_; k0 += 32) {
        bf16x16 a_nxt;
        bf16x16 b_nxt[4];
        if (k0 + 32 < C_) {                       // issue next-step loads first
            a_nxt = load_a_frag(A, row, C_, k0 + 32);
#pragma unroll
            for (int t = 0; t < 4; ++t)
                b_nxt[t] = load_b_frag_kcont(Wt, k0 + 32, n0 + 16 * t, C_);
        }
#pragma unroll
        for (int t = 0; t < 4; ++t) acc[t] = wmma_bf16(a_cur, b_cur[t], acc[t]);
        a_cur = a_nxt;
#pragma unroll
        for (int t = 0; t < 4; ++t) b_cur[t] = b_nxt[t];
    }
}

// ---------------------------------------------------------------------------
// 1a) elementwise add + f32 -> bf16 convert (activations)
// ---------------------------------------------------------------------------
__global__ void k_addcvt(const float* __restrict__ a, const float* __restrict__ b,
                         bf16_t* __restrict__ out, int n) {
    int i = blockIdx.x * blockDim.x + threadIdx.x;
    if (i < n) {
        float v = a[i] + (b ? b[i] : 0.0f);
        out[i] = f2bf(v);
    }
}

// 1b) weight convert + transpose: W[k][n] f32 -> Wt[n][k] bf16  (256x256)
__global__ void k_cvtw(const float* __restrict__ w, bf16_t* __restrict__ wt) {
    int i = blockIdx.x * blockDim.x + threadIdx.x;   // over C_*C_
    int k = i >> 8, n = i & (C_ - 1);
    wt[(size_t)n * C_ + k] = f2bf(w[i]);
}

// ---------------------------------------------------------------------------
// 2) separable RPE MLP: rpe_d[b,q,pos,h] = relu(delta @ w1 + b1) @ w2
//    One thread per (b,q,pos); weights staged in LDS.
// ---------------------------------------------------------------------------
__global__ void k_rpe(const float* __restrict__ refpts,
                      const float* __restrict__ w1, const float* __restrict__ b1,
                      const float* __restrict__ w2, float* __restrict__ out, int is_y) {
    __shared__ float w1s[2 * RPE_HID_];
    __shared__ float b1s[RPE_HID_];
    __shared__ float w2s[RPE_HID_ * NH_];
    for (int i = threadIdx.x; i < 2 * RPE_HID_;  i += blockDim.x) w1s[i] = w1[i];
    for (int i = threadIdx.x; i < RPE_HID_;      i += blockDim.x) b1s[i] = b1[i];
    for (int i = threadIdx.x; i < RPE_HID_*NH_;  i += blockDim.x) w2s[i] = w2[i];
    __syncthreads();

    int id = blockIdx.x * blockDim.x + threadIdx.x;   // over B*NQ*64
    if (id >= B_ * NQ_ * W_) return;
    int x = id % W_;
    int q = (id / W_) % NQ_;
    int b = id / (W_ * NQ_);

    const float* rp = refpts + (size_t)(b * NQ_ + q) * 4;
    float ctr = is_y ? rp[1] : rp[0];
    float sz  = is_y ? rp[3] : rp[2];
    float lo  = (ctr - sz * 0.5f) * (W_ * STRIDE_);   // W*stride == H*stride == 1024
    float hi  = (ctr + sz * 0.5f) * (W_ * STRIDE_);
    float pos = ((float)x + 0.5f) * STRIDE_;
    float d0 = lo - pos, d1 = hi - pos;

    float acc[NH_];
#pragma unroll
    for (int h = 0; h < NH_; ++h) acc[h] = 0.0f;
    for (int j = 0; j < RPE_HID_; ++j) {
        float hv = fmaxf(d0 * w1s[j] + d1 * w1s[RPE_HID_ + j] + b1s[j], 0.0f);
#pragma unroll
        for (int h = 0; h < NH_; ++h) acc[h] += hv * w2s[j * NH_ + h];
    }
    float* o = out + (size_t)id * NH_;
#pragma unroll
    for (int h = 0; h < NH_; ++h) o[h] = acc[h];
}

// ---------------------------------------------------------------------------
// 3) bf16 WMMA GEMM: X(M,256) @ W(256,256) + bias, scaled.  One wave = 16x64.
//    SEQ is a template constant -> compile-time div/mod in the epilogue.
//    TRANSV=false: Out[b][h][s][d]   (Q, K)
//    TRANSV=true : Out[b][h][d][s]   (V, transposed for contiguous P.V loads)
// ---------------------------------------------------------------------------
template <int SEQ, bool TRANSV>
__global__ void __launch_bounds__(32)
k_gemm_qkv(const bf16_t* __restrict__ A, const bf16_t* __restrict__ Wt,
           const float* __restrict__ bias, bf16_t* __restrict__ Out,
           int M, float scale) {
    int lane = threadIdx.x & 31;
    int m0 = blockIdx.x * 16;
    int n0 = blockIdx.y * 64;
    int row = m0 + (lane & 15);
    if (row >= M) row = M - 1;                 // clamp (no EXEC masking for WMMA)

    f32x8 acc[4];
    gemm_16x64(A, row, Wt, n0, acc);

    int nlane = lane & 15;
    int half8 = (lane >> 4) * 8;
#pragma unroll
    for (int t = 0; t < 4; ++t) {
        int n = n0 + 16 * t + nlane;
        int h = n >> 5, d = n & 31;
        float bv = bias[n];
#pragma unroll
        for (int j = 0; j < 8; ++j) {
            int mg = m0 + half8 + j;           // D layout: row = j + 8*(lane>>4)
            if (mg < M) {
                int bb = mg / SEQ, s = mg % SEQ;   // compile-time SEQ -> mul/shift
                float v = (acc[t][j] + bv) * scale;
                size_t idx = TRANSV
                    ? (((size_t)bb * NH_ + h) * HD_ + d) * SEQ + s
                    : (((size_t)bb * NH_ + h) * SEQ + s) * HD_ + d;
                Out[idx] = f2bf(v);
            }
        }
    }
}

// ---------------------------------------------------------------------------
// 4) output projection: Xo(600,256) @ Wp + bp -> f32 out (row-major)
// ---------------------------------------------------------------------------
__global__ void __launch_bounds__(32)
k_gemm_out(const bf16_t* __restrict__ A, const bf16_t* __restrict__ Wt,
           const float* __restrict__ bias, float* __restrict__ Out, int M) {
    int lane = threadIdx.x & 31;
    int m0 = blockIdx.x * 16;
    int n0 = blockIdx.y * 64;
    int row = m0 + (lane & 15);
    if (row >= M) row = M - 1;

    f32x8 acc[4];
    gemm_16x64(A, row, Wt, n0, acc);

    int nlane = lane & 15;
    int half8 = (lane >> 4) * 8;
#pragma unroll
    for (int t = 0; t < 4; ++t) {
        int n = n0 + 16 * t + nlane;
        float bv = bias[n];
#pragma unroll
        for (int j = 0; j < 8; ++j) {
            int mg = m0 + half8 + j;
            if (mg < M) Out[(size_t)mg * C_ + n] = acc[t][j] + bv;
        }
    }
}

// ---------------------------------------------------------------------------
// 5) fused attention: S = Q K^T + rpe_x + rpe_y + mask, online softmax, P V.
//    One wave per (q-tile, b*h, key-split).  Writes unnormalized O + (m, l).
//    K is [bh][key][d]; V is transposed [bh][d][key] -> all b128 loads.
//    V fragments are loaded BEFORE the softmax so their latency hides under
//    the shuffle/exp VALU work.
// ---------------------------------------------------------------------------
__global__ void __launch_bounds__(32)
k_attn(const bf16_t* __restrict__ Qh, const bf16_t* __restrict__ Kh,
       const bf16_t* __restrict__ Vt, const float* __restrict__ rpeX,
       const float* __restrict__ rpeY, const float* __restrict__ mask,
       float* __restrict__ Opart, float* __restrict__ Mpart,
       float* __restrict__ Lpart) {
    __shared__ bf16_t Pls[16 * 32];            // P tile, D-layout -> A-layout transpose

    int lane  = threadIdx.x & 31;
    int qt    = blockIdx.x;
    int bh    = blockIdx.y;
    int sp    = blockIdx.z;
    int b     = bh >> 3, h = bh & 7;
    int nlane = lane & 15;
    int half8 = (lane >> 4) * 8;

    // Q fragment (HD=32 == WMMA K exactly), rows clamped into [0, NQ)
    int qrow = qt * 16 + nlane; if (qrow >= NQ_) qrow = NQ_ - 1;
    bf16x16 aQ = load_a_frag(Qh + (size_t)bh * NQ_ * HD_, qrow, HD_, 0);

    float mrun[8], lrun[8];
    f32x8 o0, o1;
#pragma unroll
    for (int j = 0; j < 8; ++j) { mrun[j] = -3.0e38f; lrun[j] = 0.0f; o0[j] = 0.0f; o1[j] = 0.0f; }

    const bf16_t* Kb  = Kh + (size_t)bh * HW_ * HD_;
    const bf16_t* VtB = Vt + (size_t)bh * HD_ * HW_;

    for (int c = 0; c < KEYS_PER_SPLIT_ / 32; ++c) {
        int key0 = sp * KEYS_PER_SPLIT_ + c * 32;

        // prefetch next chunk of K and V while this one computes
        if (c + 1 < KEYS_PER_SPLIT_ / 32) {
            __builtin_prefetch(Kb + (size_t)(key0 + 32 + lane) * HD_, 0, 3);
            __builtin_prefetch(VtB + (size_t)lane * HW_ + key0 + 32, 0, 3);
        }

        // ---- scores: two 16x16 WMMA tiles over 32 keys (K^T is k-contiguous)
        bf16x16 bK0 = load_b_frag_kcont(Kb, 0, key0, HD_);        // keys key0..+15
        bf16x16 bK1 = load_b_frag_kcont(Kb, 0, key0 + 16, HD_);   // keys +16..+31
        // V fragments issued now; consumed after softmax (latency hidden)
        bf16x16 bV0 = load_b_frag_kcont(VtB, key0, 0, HW_);       // dims 0..15
        bf16x16 bV1 = load_b_frag_kcont(VtB, key0, 16, HW_);      // dims 16..31

        f32x8 s0, s1;
#pragma unroll
        for (int j = 0; j < 8; ++j) { s0[j] = 0.0f; s1[j] = 0.0f; }
        s0 = wmma_bf16(aQ, bK0, s0);
        s1 = wmma_bf16(aQ, bK1, s1);

        // ---- separable rpe + padding mask (key row y constant within chunk)
        int y  = key0 / W_;
        int x0 = (key0 % W_) + nlane;
        float mk0 = mask[b * HW_ + key0 + nlane] * -100.0f;
        float mk1 = mask[b * HW_ + key0 + 16 + nlane] * -100.0f;
#pragma unroll
        for (int j = 0; j < 8; ++j) {
            int qg = qt * 16 + half8 + j; if (qg >= NQ_) qg = NQ_ - 1;
            size_t qb = (size_t)(b * NQ_ + qg);
            float ry  = rpeY[(qb * H_ + y) * NH_ + h];
            float rx0 = rpeX[(qb * W_ + x0) * NH_ + h];
            float rx1 = rpeX[(qb * W_ + x0 + 16) * NH_ + h];
            s0[j] += ry + rx0 + mk0;
            s1[j] += ry + rx1 + mk1;
        }

        // ---- online softmax: row = j + 8*(lane>>4); keys live across 16 lanes
#pragma unroll
        for (int j = 0; j < 8; ++j) {
            float rm = fmaxf(s0[j], s1[j]);
            rm = fmaxf(rm, __shfl_xor(rm, 1, 32));
            rm = fmaxf(rm, __shfl_xor(rm, 2, 32));
            rm = fmaxf(rm, __shfl_xor(rm, 4, 32));
            rm = fmaxf(rm, __shfl_xor(rm, 8, 32));
            float mn    = fmaxf(mrun[j], rm);
            float alpha = __expf(mrun[j] - mn);
            float p0 = __expf(s0[j] - mn);
            float p1 = __expf(s1[j] - mn);
            float rs = p0 + p1;
            rs += __shfl_xor(rs, 1, 32);
            rs += __shfl_xor(rs, 2, 32);
            rs += __shfl_xor(rs, 4, 32);
            rs += __shfl_xor(rs, 8, 32);
            lrun[j] = lrun[j] * alpha + rs;
            mrun[j] = mn;
            o0[j] *= alpha; o1[j] *= alpha;
            Pls[(half8 + j) * 32 + nlane]      = f2bf(p0);
            Pls[(half8 + j) * 32 + 16 + nlane] = f2bf(p1);
        }
        __syncthreads();

        // ---- P (A-layout from LDS) x V
        bf16x16 aP;
        {
            const bf16_t* pb = &Pls[nlane * 32 + ((lane >> 4) << 3)];
#pragma unroll
            for (int e = 0; e < 8; ++e) { aP[e] = pb[e]; aP[8 + e] = pb[16 + e]; }
        }
        o0 = wmma_bf16(aP, bV0, o0);
        o1 = wmma_bf16(aP, bV1, o1);
        __syncthreads();
    }

    // ---- write per-split partials (unnormalized numerator + running m, l)
#pragma unroll
    for (int j = 0; j < 8; ++j) {
        int ql = qt * 16 + half8 + j;          // < NQPAD_
        size_t ob = ((size_t)(bh * NQPAD_ + ql) * NSPLIT_ + sp) * HD_;
        Opart[ob + nlane]      = o0[j];
        Opart[ob + 16 + nlane] = o1[j];
        if (nlane == 0) {
            Mpart[(size_t)(bh * NQPAD_ + ql) * NSPLIT_ + sp] = mrun[j];
            Lpart[(size_t)(bh * NQPAD_ + ql) * NSPLIT_ + sp] = lrun[j];
        }
    }
}

// ---------------------------------------------------------------------------
// 6) log-sum-exp combine across key-splits -> Xo bf16 (b, q, h*HD + d)
// ---------------------------------------------------------------------------
__global__ void __launch_bounds__(32)
k_combine(const float* __restrict__ Opart, const float* __restrict__ Mpart,
          const float* __restrict__ Lpart, bf16_t* __restrict__ Xo) {
    int bid = blockIdx.x;                      // over 16 * NQ_
    int bh  = bid / NQ_;
    int q   = bid % NQ_;
    int d   = threadIdx.x;                     // 0..31 == HD
    size_t mlbase = (size_t)(bh * NQPAD_ + q) * NSPLIT_;

    float mstar = -3.0e38f;
    for (int sp = 0; sp < NSPLIT_; ++sp) mstar = fmaxf(mstar, Mpart[mlbase + sp]);
    float lsum = 0.0f, osum = 0.0f;
    for (int sp = 0; sp < NSPLIT_; ++sp) {
        float w = __expf(Mpart[mlbase + sp] - mstar);
        lsum += w * Lpart[mlbase + sp];
        osum += w * Opart[(mlbase + sp) * HD_ + d];
    }
    int b = bh >> 3, h = bh & 7;
    Xo[((size_t)(b * NQ_ + q)) * C_ + h * HD_ + d] = f2bf(osum / lsum);
}

// ---------------------------------------------------------------------------
// host launcher
// ---------------------------------------------------------------------------
extern "C" void kernel_launch(void* const* d_in, const int* in_sizes, int n_in,
                              void* d_out, int out_size, void* d_ws, size_t ws_size,
                              hipStream_t stream) {
    const float* raw_query = (const float*)d_in[0];
    const float* query_pos = (const float*)d_in[1];
    const float* refpts    = (const float*)d_in[2];
    const float* raw_src   = (const float*)d_in[3];
    const float* src_pos   = (const float*)d_in[4];
    const float* padmask   = (const float*)d_in[5];
    // d_in[6] = input_spatial_shapes (64x64, baked in as constants)
    const float* Wq = (const float*)d_in[7];
    const float* bq = (const float*)d_in[8];
    const float* Wk = (const float*)d_in[9];
    const float* bk = (const float*)d_in[10];
    const float* Wv = (const float*)d_in[11];
    const float* bv = (const float*)d_in[12];
    const float* Wp = (const float*)d_in[13];
    const float* bp = (const float*)d_in[14];
    const float* c1w1 = (const float*)d_in[15];
    const float* c1b1 = (const float*)d_in[16];
    const float* c1w2 = (const float*)d_in[17];
    const float* c2w1 = (const float*)d_in[18];
    const float* c2b1 = (const float*)d_in[19];
    const float* c2w2 = (const float*)d_in[20];

    const int MQ = B_ * NQ_;     // 600
    const int MS = B_ * HW_;     // 8192

    // workspace carve-up (~26 MB total)
    char* ws = (char*)d_ws;
    size_t off = 0;
    auto carve = [&](size_t bytes) -> void* {
        void* p = ws + off;
        off += (bytes + 255) & ~(size_t)255;
        return p;
    };
    bf16_t* Xq  = (bf16_t*)carve((size_t)MQ * C_ * 2);
    bf16_t* Xs  = (bf16_t*)carve((size_t)MS * C_ * 2);
    bf16_t* Xv  = (bf16_t*)carve((size_t)MS * C_ * 2);
    bf16_t* tWq = (bf16_t*)carve((size_t)C_ * C_ * 2);
    bf16_t* tWk = (bf16_t*)carve((size_t)C_ * C_ * 2);
    bf16_t* tWv = (bf16_t*)carve((size_t)C_ * C_ * 2);
    bf16_t* tWp = (bf16_t*)carve((size_t)C_ * C_ * 2);
    bf16_t* Qh  = (bf16_t*)carve((size_t)B_ * NH_ * NQ_ * HD_ * 2);
    bf16_t* Kh  = (bf16_t*)carve((size_t)B_ * NH_ * HW_ * HD_ * 2);
    bf16_t* Vt  = (bf16_t*)carve((size_t)B_ * NH_ * HW_ * HD_ * 2);
    float*  rX  = (float*) carve((size_t)B_ * NQ_ * W_ * NH_ * 4);
    float*  rY  = (float*) carve((size_t)B_ * NQ_ * H_ * NH_ * 4);
    float*  Op  = (float*) carve((size_t)B_ * NH_ * NQPAD_ * NSPLIT_ * HD_ * 4);
    float*  Mp  = (float*) carve((size_t)B_ * NH_ * NQPAD_ * NSPLIT_ * 4);
    float*  Lp  = (float*) carve((size_t)B_ * NH_ * NQPAD_ * NSPLIT_ * 4);
    bf16_t* Xo  = (bf16_t*)carve((size_t)MQ * C_ * 2);
    (void)ws_size; (void)in_sizes; (void)n_in; (void)out_size;

    // 1) bf16 staging (activations) + weight convert/transpose
    k_addcvt<<<(MQ * C_ + 255) / 256, 256, 0, stream>>>(raw_query, query_pos, Xq, MQ * C_);
    k_addcvt<<<(MS * C_ + 255) / 256, 256, 0, stream>>>(raw_src, src_pos, Xs, MS * C_);
    k_addcvt<<<(MS * C_ + 255) / 256, 256, 0, stream>>>(raw_src, nullptr, Xv, MS * C_);
    k_cvtw<<<(C_ * C_) / 256, 256, 0, stream>>>(Wq, tWq);
    k_cvtw<<<(C_ * C_) / 256, 256, 0, stream>>>(Wk, tWk);
    k_cvtw<<<(C_ * C_) / 256, 256, 0, stream>>>(Wv, tWv);
    k_cvtw<<<(C_ * C_) / 256, 256, 0, stream>>>(Wp, tWp);

    // 2) separable RPE MLPs
    k_rpe<<<(B_ * NQ_ * W_ + 255) / 256, 256, 0, stream>>>(refpts, c1w1, c1b1, c1w2, rX, 0);
    k_rpe<<<(B_ * NQ_ * H_ + 255) / 256, 256, 0, stream>>>(refpts, c2w1, c2b1, c2w2, rY, 1);

    // 3) Q/K/V projections (bf16 WMMA); V written transposed
    const float qscale = 0.17677669529663687f;   // 1/sqrt(32)
    k_gemm_qkv<NQ_, false><<<dim3((MQ + 15) / 16, C_ / 64), 32, 0, stream>>>(Xq, tWq, bq, Qh, MQ, qscale);
    k_gemm_qkv<HW_, false><<<dim3(MS / 16, C_ / 64), 32, 0, stream>>>(Xs, tWk, bk, Kh, MS, 1.0f);
    k_gemm_qkv<HW_, true ><<<dim3(MS / 16, C_ / 64), 32, 0, stream>>>(Xv, tWv, bv, Vt, MS, 1.0f);

    // 4) fused attention (split-K over keys) + combine
    k_attn<<<dim3((NQ_ + 15) / 16, B_ * NH_, NSPLIT_), 32, 0, stream>>>(
        Qh, Kh, Vt, rX, rY, padmask, Op, Mp, Lp);
    k_combine<<<B_ * NH_ * NQ_, 32, 0, stream>>>(Op, Mp, Lp, Xo);

    // 5) output projection (f32 out)
    k_gemm_out<<<dim3((MQ + 15) / 16, C_ / 64), 32, 0, stream>>>(Xo, tWp, bp, (float*)d_out, MQ);
}
